// GraphAttentionLayer_26920855011723
// MI455X (gfx1250) — matrix operand order repaired and checked by reading, not compiled, not run on hardware
//
#include <hip/hip_runtime.h>
#include <hip/hip_bf16.h>
#include <stdint.h>

#define NROWS 8192
#define IN_F  512
#define OUT_F 256

typedef __attribute__((ext_vector_type(16))) __bf16 v16bf;
typedef __attribute__((ext_vector_type(8)))  float  v8f;

union Frag16 { v16bf v; uint32_t u[8]; uint4 q[2]; };

// A-fragment k index for 16x32 bf16 A (ISA 7.12.2): lane half (0/1), v = 0..15
__device__ __forceinline__ int afrag_k(int hf, int v) {
    return (v < 8) ? (hf * 8 + v) : (8 + hf * 8 + v);
}

// ---------------------------------------------------------------------------
// K0: convert X -> bf16 (row major) and W -> bf16 transposed (WbT[n][k])
// ---------------------------------------------------------------------------
__global__ void k_convert(const float* __restrict__ X, const float* __restrict__ W,
                          __bf16* __restrict__ Xb, __bf16* __restrict__ WbT) {
    int i = blockIdx.x * blockDim.x + threadIdx.x;
    const int nX = NROWS * IN_F;
    if (i < nX) {
        Xb[i] = (__bf16)X[i];
    } else {
        int j = i - nX;
        if (j < IN_F * OUT_F) {
            int n = j % OUT_F, k = j / OUT_F;
            WbT[n * IN_F + k] = (__bf16)W[k * OUT_F + n];
        }
    }
}

// ---------------------------------------------------------------------------
// K1: Wh = X @ W via WMMA bf16. One wave computes a 16x16 tile.
// Writes Wh (f32, row major) and WhT (bf16, [OUT_F][NROWS]) for GEMM2 B-frags.
// ---------------------------------------------------------------------------
__global__ void __launch_bounds__(256) k_gemm1(const __bf16* __restrict__ Xb,
                                               const __bf16* __restrict__ WbT,
                                               float* __restrict__ Wh,
                                               __bf16* __restrict__ WhT) {
    int lane = threadIdx.x & 31;
    int wv   = threadIdx.x >> 5;
    int tile = blockIdx.x * 8 + wv;              // 512 row-tiles x 16 col-tiles
    int m0   = (tile >> 4) << 4;
    int n0   = (tile & 15) << 4;
    int r    = lane & 15, hf = lane >> 4;

    v8f acc = {0.f, 0.f, 0.f, 0.f, 0.f, 0.f, 0.f, 0.f};
    const uint32_t* Arow = (const uint32_t*)(Xb  + (size_t)(m0 + r) * IN_F);
    const uint32_t* Brow = (const uint32_t*)(WbT + (size_t)(n0 + r) * IN_F);

    for (int k0 = 0; k0 < IN_F; k0 += 32) {
        Frag16 a, b;
#pragma unroll
        for (int vg = 0; vg < 8; ++vg) {
            a.u[vg] = Arow[(k0 + afrag_k(hf, 2 * vg)) >> 1];
            b.u[vg] = Brow[(k0 + afrag_k(hf, 2 * vg)) >> 1];
        }
        acc = __builtin_amdgcn_wmma_f32_16x16x32_bf16(false, a.v, false, b.v,
                                                      (short)0, acc, false, false);
    }
#pragma unroll
    for (int jj = 0; jj < 8; ++jj) {
        int m = m0 + hf * 8 + jj;
        float val = acc[jj];
        Wh [(size_t)m * OUT_F + n0 + r]  = val;
        WhT[(size_t)(n0 + r) * NROWS + m] = (__bf16)val;
    }
}

// ---------------------------------------------------------------------------
// K2: Wh1[i] = Wh[i,:] . W2[0:256],  Wh2[i] = Wh[i,:] . W2[256:512]
// One wave per row, shuffle reduction (wave32).
// ---------------------------------------------------------------------------
__global__ void __launch_bounds__(256) k_wh12(const float* __restrict__ Wh,
                                              const float* __restrict__ W2,
                                              float* __restrict__ Wh1,
                                              float* __restrict__ Wh2) {
    int lane = threadIdx.x & 31;
    int wv   = threadIdx.x >> 5;
    int row  = blockIdx.x * 8 + wv;
    const float* wrow = Wh + (size_t)row * OUT_F + lane * 8;
    float d1 = 0.f, d2 = 0.f;
#pragma unroll
    for (int e = 0; e < 8; ++e) {
        float v = wrow[e];
        d1 += v * W2[lane * 8 + e];
        d2 += v * W2[OUT_F + lane * 8 + e];
    }
#pragma unroll
    for (int off = 16; off > 0; off >>= 1) {
        d1 += __shfl_xor(d1, off, 32);
        d2 += __shfl_xor(d2, off, 32);
    }
    if (lane == 0) { Wh1[row] = d1; Wh2[row] = d2; }
}

// ---------------------------------------------------------------------------
// K3: fused attention. Block = 256 threads (8 waves) owns 16 output rows.
// Phase 1: row sums of exp(e) (masked entries contribute exp(0)=1; fixed max=0
//          is numerically safe here, identical math to reference softmax).
// Phase 2: flash-style P@Wh with bf16 WMMA; P tile staged in LDS in the exact
//          A-fragment per-lane order, each wave handles 2 column tiles.
// ---------------------------------------------------------------------------
__global__ void __launch_bounds__(256) k_attn(const int* __restrict__ adj,
                                              const float* __restrict__ Wh1,
                                              const float* __restrict__ Wh2,
                                              const __bf16* __restrict__ WhT,
                                              float* __restrict__ out) {
    __shared__ __align__(16) __bf16 ldsP[512];   // 16x32 P tile in A-frag order
    __shared__ float ps[256];
    __shared__ float wh1s[16];
    __shared__ float invl[16];

    int t = threadIdx.x;
    int i0 = blockIdx.x * 16;
    if (t < 16) wh1s[t] = Wh1[i0 + t];
    __syncthreads();

    int prow = t >> 4;                 // 0..15 : local row this thread scans
    int pv   = t & 15;
    const int* arow = adj + (size_t)(i0 + prow) * NROWS;
    float w1 = wh1s[prow];

    // ---- phase 1: softmax denominators ----
    {
        float s = 0.f;
        for (int j = pv; j < NROWS; j += 16) {
            float e = w1 + Wh2[j];
            e = e > 0.f ? e : 0.2f * e;          // leaky_relu(0.2)
            s += (arow[j] > 0) ? __expf(e) : 1.0f; // masked -> exp(0) = 1
        }
        ps[t] = s;
    }
    __syncthreads();
    if (t < 16) {
        float s = 0.f;
#pragma unroll
        for (int c = 0; c < 16; ++c) s += ps[t * 16 + c];
        invl[t] = 1.0f / s;
    }
    __syncthreads();

    // ---- phase 2: O = P @ Wh ----
    int lane = t & 31;
    int wv   = t >> 5;
    int r    = lane & 15, hf = lane >> 4;
    int n0a  = wv * 32, n0b = wv * 32 + 16;      // 8 waves x 2 tiles = 256 cols
    v8f accA = {0.f, 0.f, 0.f, 0.f, 0.f, 0.f, 0.f, 0.f};
    v8f accB = {0.f, 0.f, 0.f, 0.f, 0.f, 0.f, 0.f, 0.f};
    const uint32_t* BrowA = (const uint32_t*)(WhT + (size_t)(n0a + r) * NROWS);
    const uint32_t* BrowB = (const uint32_t*)(WhT + (size_t)(n0b + r) * NROWS);

    // this thread's two P elements: A-frag slots (l=prow, half 0) and (half 1)
    int pk0 = afrag_k(0, pv);
    int pk1 = afrag_k(1, pv);

    for (int j0 = 0; j0 < NROWS; j0 += 32) {
        __builtin_prefetch(arow + j0 + 512, 0, 0);   // global_prefetch_b8
        {
            float e0 = w1 + Wh2[j0 + pk0];
            e0 = e0 > 0.f ? e0 : 0.2f * e0;
            float p0 = (arow[j0 + pk0] > 0) ? __expf(e0) : 1.0f;
            float e1 = w1 + Wh2[j0 + pk1];
            e1 = e1 > 0.f ? e1 : 0.2f * e1;
            float p1 = (arow[j0 + pk1] > 0) ? __expf(e1) : 1.0f;
            ldsP[t]       = (__bf16)p0;   // idx = prow*16 + pv
            ldsP[t + 256] = (__bf16)p1;   // idx = (prow+16)*16 + pv
        }
        __syncthreads();

        Frag16 a;
        const uint4* pa = reinterpret_cast<const uint4*>(&ldsP[(size_t)lane * 16]);
        a.q[0] = pa[0];
        a.q[1] = pa[1];

        Frag16 ba, bb;
#pragma unroll
        for (int vg = 0; vg < 8; ++vg) {
            int k = j0 + afrag_k(hf, 2 * vg);
            ba.u[vg] = BrowA[k >> 1];
            bb.u[vg] = BrowB[k >> 1];
        }
        accA = __builtin_amdgcn_wmma_f32_16x16x32_bf16(false, a.v, false, ba.v,
                                                       (short)0, accA, false, false);
        accB = __builtin_amdgcn_wmma_f32_16x16x32_bf16(false, a.v, false, bb.v,
                                                       (short)0, accB, false, false);
        __syncthreads();
    }

#pragma unroll
    for (int jj = 0; jj < 8; ++jj) {
        int lr = hf * 8 + jj;
        int m  = i0 + lr;
        float sA = accA[jj] * invl[lr];
        float sB = accB[jj] * invl[lr];
        out[(size_t)m * OUT_F + n0a + r] = sA > 0.f ? sA : 0.f;   // relu
        out[(size_t)m * OUT_F + n0b + r] = sB > 0.f ? sB : 0.f;
    }
}

// ---------------------------------------------------------------------------
extern "C" void kernel_launch(void* const* d_in, const int* in_sizes, int n_in,
                              void* d_out, int out_size, void* d_ws, size_t ws_size,
                              hipStream_t stream) {
    const float* X   = (const float*)d_in[0];   // [8192,512]
    const float* W   = (const float*)d_in[1];   // [512,256]
    const float* W2  = (const float*)d_in[2];   // [512,1]
    const int*   adj = (const int*)d_in[3];     // [8192,8192]
    float* out = (float*)d_out;                 // [8192,256]

    char* ws = (char*)d_ws;
    size_t o = 0;
    __bf16* Xb  = (__bf16*)(ws + o); o += (size_t)NROWS * IN_F * 2;   // 8 MB
    __bf16* WbT = (__bf16*)(ws + o); o += (size_t)IN_F * OUT_F * 2;   // 256 KB
    float*  Wh  = (float*)(ws + o);  o += (size_t)NROWS * OUT_F * 4;  // 8 MB
    __bf16* WhT = (__bf16*)(ws + o); o += (size_t)OUT_F * NROWS * 2;  // 4 MB
    float*  Wh1 = (float*)(ws + o);  o += (size_t)NROWS * 4;
    float*  Wh2 = (float*)(ws + o);  o += (size_t)NROWS * 4;
    (void)ws_size; (void)in_sizes; (void)n_in; (void)out_size;

    int convN = NROWS * IN_F + IN_F * OUT_F;
    k_convert<<<(convN + 255) / 256, 256, 0, stream>>>(X, W, Xb, WbT);
    k_gemm1 <<<(NROWS / 16) * (OUT_F / 16) / 8, 256, 0, stream>>>(Xb, WbT, Wh, WhT);
    k_wh12  <<<NROWS / 8, 256, 0, stream>>>(Wh, W2, Wh1, Wh2);
    k_attn  <<<NROWS / 16, 256, 0, stream>>>(adj, Wh1, Wh2, WhT, out);
}
